// KANshared_3607772529057
// MI455X (gfx1250) — compile-verified, matrix operand order checked by reading, not run on hardware
//
#include <hip/hip_runtime.h>

typedef __attribute__((ext_vector_type(16))) _Float16 v16h;
typedef __attribute__((ext_vector_type(8)))  float    v8f;

#define SEQ       65536
#define IN_DIM    64
#define OUT_DIM   128
#define BM        128          // rows (s positions) per block
#define KTOT      256          // 64 features * degrees 1..4 (degree 0 folded into bias)
#define BSTRIDE_H 280          // B_lds row stride in halves: 140 dwords == 12 mod 64 -> conflict-free b128 reads
#define CSTRIDE_F 129          // C_lds row stride in floats

union __align__(16) SMem {
    _Float16 b[OUT_DIM * BSTRIDE_H];  // 71680 bytes: B^T [n][k] in f16
    float    c[OUT_DIM * CSTRIDE_F];  // 66048 bytes: output transpose staging
};

// CDNA5 hardware tanh (TRANS32 op). Builtin if declared, else inline asm with the
// required 1-op gap after a TRANS instruction baked in.
static __device__ __forceinline__ float fast_tanh(float v) {
#if __has_builtin(__builtin_amdgcn_tanhf)
    return __builtin_amdgcn_tanhf(v);
#else
    float r;
    asm volatile("v_tanh_f32 %0, %1\n\tv_nop" : "=v"(r) : "v"(v));
    return r;
#endif
}

__global__ __launch_bounds__(256)
void kan_jacobi_wmma(const float* __restrict__ x,
                     const float* __restrict__ coeffs,
                     float* __restrict__ y)
{
    __shared__ SMem  smem;
    __shared__ float s_bias[OUT_DIM];

    const int tid  = threadIdx.x;
    const int lane = tid & 31;
    const int w    = tid >> 5;        // wave 0..7 -> 16-row strip
    const int lrow = lane & 15;       // A row / B,C column within tile
    const int hi   = lane >> 4;       // K-offset group per ISA fragment layout

    const int rowbase = blockIdx.x * BM;          // global row = b*SEQ + s
    const int b       = rowbase >> 16;            // SEQ == 65536, BM divides SEQ
    const int sbase   = rowbase & (SEQ - 1);

    // ---- Stage B^T (f16) and per-output bias (degree-0 term) into LDS ----
    // k = i*4 + (d-1); coeffs layout (i, o, d) with strides (640, 5, 1)
    for (int idx = tid; idx < OUT_DIM * KTOT; idx += 256) {
        int n = idx >> 8;             // / KTOT
        int k = idx & (KTOT - 1);
        int i = k >> 2;
        int d = (k & 3) + 1;
        smem.b[n * BSTRIDE_H + k] = (_Float16)coeffs[i * (OUT_DIM * 5) + n * 5 + d];
    }
    if (tid < OUT_DIM) {
        float s = 0.f;
        for (int i = 0; i < IN_DIM; ++i)
            s += coeffs[i * (OUT_DIM * 5) + tid * 5 + 0];   // P0 == 1 -> bias
        s_bias[tid] = s;
    }
    __syncthreads();

    // ---- Init accumulators with bias (C layout: column o constant per lane) ----
    v8f acc[8];
#pragma unroll
    for (int nt = 0; nt < 8; ++nt) {
        float bv = s_bias[nt * 16 + lrow];
        v8f t = {bv, bv, bv, bv, bv, bv, bv, bv};
        acc[nt] = t;
    }

    // ---- Main loop: A fragment computed in-register (tanh + Jacobi recurrence) ----
    const int    srow = sbase + w * 16 + lrow;    // this lane's s position
    const size_t xrow = ((size_t)(b * IN_DIM) << 16) + (size_t)srow;

    // alpha = beta = 1 Jacobi recurrence: P1 = 2x; P2 = 1.875x*P1 - 0.75;
    // P3 = (56/30)x*P2 - 0.8*P1; P4 = 1.875x*P3 - (5/6)*P2
    auto jac4 = [&](int i, int base, v16h& a) {
        float xt = fast_tanh(x[xrow + ((size_t)i << 16)]);
        float p1 = 2.0f * xt;
        float p2 = 1.875f * xt * p1 - 0.75f;
        float p3 = 1.8666666666666667f * xt * p2 - 0.8f * p1;
        float p4 = 1.875f * xt * p3 - 0.8333333333333333f * p2;
        a[base + 0] = (_Float16)p1;
        a[base + 1] = (_Float16)p2;
        a[base + 2] = (_Float16)p3;
        a[base + 3] = (_Float16)p4;
    };

#pragma unroll
    for (int kc = 0; kc < KTOT / 32; ++kc) {
        const int k0 = kc * 32;
        const int i0 = kc * 8 + hi * 2;           // fragment K mapping per ISA table
        v16h a;
        jac4(i0,     0,  a);                      // K = k0 + hi*8 + 0..7   (v[0..3])
        jac4(i0 + 1, 4,  a);
        jac4(i0 + 4, 8,  a);                      // K = k0 + 16 + hi*8 + 0..7 (v[4..7])
        jac4(i0 + 5, 12, a);

        const _Float16* brow = smem.b + (size_t)(k0 + hi * 8);
#pragma unroll
        for (int nt = 0; nt < 8; ++nt) {
            const _Float16* bp = brow + (size_t)(nt * 16 + lrow) * BSTRIDE_H;
            v16h bf;
            uint4 u0 = *(const uint4*)bp;         // K seg 0 (16B aligned: 560B row stride)
            uint4 u1 = *(const uint4*)(bp + 16);  // K seg 1
            __builtin_memcpy(&bf, &u0, 16);
            __builtin_memcpy((char*)&bf + 16, &u1, 16);
            acc[nt] = __builtin_amdgcn_wmma_f32_16x16x32_f16(
                false, a, false, bf, (short)0, acc[nt], false, false);
        }
    }

    // ---- Transpose C through LDS (reuses B region) for s-contiguous stores ----
    __syncthreads();                              // all waves done reading smem.b
#pragma unroll
    for (int nt = 0; nt < 8; ++nt) {
        int o = nt * 16 + lrow;
#pragma unroll
        for (int j = 0; j < 8; ++j) {
            int sl = w * 16 + hi * 8 + j;         // C layout: lanes>=16 hold M = j+8
            smem.c[o * CSTRIDE_F + sl] = acc[nt][j];
        }
    }
    __syncthreads();

    // ---- Coalesced store: y[b][o][s], 512B contiguous per o-row ----
    const size_t obase = ((size_t)(b * OUT_DIM) << 16) + (size_t)sbase;
#pragma unroll
    for (int it = 0; it < (OUT_DIM * BM) / 256; ++it) {
        int idx = it * 256 + tid;
        int o   = idx >> 7;
        int sl  = idx & (BM - 1);
        y[obase + ((size_t)o << 16) + sl] = smem.c[o * CSTRIDE_F + sl];
    }
}

extern "C" void kernel_launch(void* const* d_in, const int* in_sizes, int n_in,
                              void* d_out, int out_size, void* d_ws, size_t ws_size,
                              hipStream_t stream) {
    (void)in_sizes; (void)n_in; (void)d_ws; (void)ws_size; (void)out_size;
    const float* x      = (const float*)d_in[0];   // (4, 64, 65536) f32
    const float* coeffs = (const float*)d_in[1];   // (64, 128, 5) f32
    float*       y      = (float*)d_out;           // (4, 128, 65536) f32
    const int rows   = 4 * SEQ;                    // 262144
    const int blocks = rows / BM;                  // 2048
    kan_jacobi_wmma<<<blocks, 256, 0, stream>>>(x, coeffs, y);
}
